// FourierKANLayer_5299989643383
// MI455X (gfx1250) — matrix operand order repaired and checked by reading, not compiled, not run on hardware
//
#include <hip/hip_runtime.h>

typedef __attribute__((ext_vector_type(16))) _Float16 v16h;
typedef __attribute__((ext_vector_type(8)))  _Float16 v8h;
typedef __attribute__((ext_vector_type(8)))  float    v8f;

#define N_ROWS 8192
#define IN_D   512
#define OUT_D  512
#define GRID_G 8
#define KDIM   (IN_D * GRID_G * 2)   /* 8192 */
#define NCHUNK (KDIM / 64)           /* 128  */

// ---------------------------------------------------------------------------
// Weight prep: fouriercoeffs[2][O][I][G] (f32) -> Wt[o][k] (f16),
// k = i*16 + g*2 + cs  (cs: 0=cos, 1=sin). One thread per (o,i) writes a
// contiguous 32B run so the GEMM's B-fragment loads are unit-stride.
// ---------------------------------------------------------------------------
__global__ __launch_bounds__(256)
void fkan_prep_weights(const float* __restrict__ fc, _Float16* __restrict__ Wt) {
    int idx = blockIdx.x * 256 + threadIdx.x;   // idx = o*IN_D + i
    int o = idx >> 9;
    int i = idx & (IN_D - 1);
    const float* cw = fc + ((size_t)o * IN_D + i) * GRID_G;              // cs = 0
    const float* sw = fc + ((size_t)(OUT_D + o) * IN_D + i) * GRID_G;    // cs = 1
    union { _Float16 h[16]; v16h v; } w;
#pragma unroll
    for (int g = 0; g < GRID_G; ++g) {
        w.h[2 * g]     = (_Float16)cw[g];
        w.h[2 * g + 1] = (_Float16)sw[g];
    }
    *(v16h*)(Wt + (size_t)o * KDIM + (size_t)i * 16) = w.v;
}

// ---------------------------------------------------------------------------
// Fused feature-gen + WMMA GEMM, double-buffered LDS.
// Block: 256 threads = 8 wave32s in a 2(M) x 4(N) grid.
// Block tile: 128(N rows) x 128(O cols). Each wave: 64x32 = 4x2 WMMA tiles.
// K loop: chunks of 64 (= 4 input dims x 8 harmonics x {cos,sin}).
// Feature gen for chunk+1 (VALU: 2 hw trans + angle-addition recurrence)
// co-executes with the XDL WMMA stream for chunk; one barrier per iteration.
// ---------------------------------------------------------------------------
__global__ __launch_bounds__(256)
void fkan_wmma(const float* __restrict__ x, const _Float16* __restrict__ Wt,
               const float* __restrict__ bias, float* __restrict__ out) {
    // +8 halves of row padding -> fragment ds_load_b128s hit all 64 banks
    __shared__ _Float16 As[2][128][72];

    const int tid    = threadIdx.x;
    const int lane   = tid & 31;
    const int wid    = tid >> 5;
    const int wave_m = wid >> 2;      // 0..1
    const int wave_n = wid & 3;       // 0..3
    const int row0   = blockIdx.y * 128;
    const int col0   = blockIdx.x * 128;
    const int lm     = lane & 15;     // M (A) or N (B, C) index within tile
    const int lk     = lane >> 4;     // K-half selector

    // ---- feature staging: 512 (n,ii) pairs per chunk, 2 per thread -------
    auto stage = [&](int chunk, int sbuf) {
        const int i0 = chunk * 4;
#pragma unroll
        for (int r = 0; r < 2; ++r) {
            int p  = tid + r * 256;
            int n  = p >> 2;          // 0..127
            int ii = p & 3;           // 0..3
            float xv = x[(size_t)(row0 + n) * IN_D + i0 + ii];
            float c1 = __cosf(xv);
            float s1 = __sinf(xv);
            union { _Float16 h[16]; v8h v[2]; } f;
            f.h[0] = (_Float16)c1;
            f.h[1] = (_Float16)s1;
            float c = c1, s = s1;
#pragma unroll
            for (int g = 1; g < GRID_G; ++g) {
                float cn = __builtin_fmaf(c, c1, -(s * s1));
                float sn = __builtin_fmaf(s, c1,  (c * s1));
                c = cn; s = sn;
                f.h[2 * g]     = (_Float16)c;
                f.h[2 * g + 1] = (_Float16)s;
            }
            v8h* dst = (v8h*)&As[sbuf][n][ii * 16];
            dst[0] = f.v[0];
            dst[1] = f.v[1];
        }
    };

    v8f acc[4][2] = {};

    stage(0, 0);
    __syncthreads();

    for (int chunk = 0; chunk < NCHUNK; ++chunk) {
        const int buf = chunk & 1;

        // Stage next chunk into the other buffer; independent of the WMMAs
        // below, so VALU/trans work overlaps the matrix pipe.
        if (chunk + 1 < NCHUNK) stage(chunk + 1, buf ^ 1);

        const _Float16* wbase = Wt + (size_t)chunk * 64;
#pragma unroll
        for (int kk = 0; kk < 64; kk += 32) {
            // B fragments: lane = column (lm), lk picks K 0-15 vs 16-31;
            // 16 contiguous halves (32B) straight from global (L2-resident).
            v16h bfrag[2];
#pragma unroll
            for (int nt = 0; nt < 2; ++nt) {
                int o = col0 + wave_n * 32 + nt * 16 + lm;
                bfrag[nt] = *(const v16h*)(wbase + (size_t)o * KDIM + kk + lk * 16);
            }
#pragma unroll
            for (int mt = 0; mt < 4; ++mt) {
                // A fragment (16-bit 16x32 layout): lane holds row lm,
                // halves = {K: kh*16..+15} (contiguous, matches B)
                int m = wave_m * 64 + mt * 16 + lm;
                v8h lo = *(const v8h*)&As[buf][m][kk + lk * 16];
                v8h hi = *(const v8h*)&As[buf][m][kk + lk * 16 + 8];
                v16h a = __builtin_shufflevector(lo, hi,
                        0, 1, 2, 3, 4, 5, 6, 7, 8, 9, 10, 11, 12, 13, 14, 15);
#pragma unroll
                for (int nt = 0; nt < 2; ++nt) {
                    acc[mt][nt] = __builtin_amdgcn_wmma_f32_16x16x32_f16(
                        /*neg_a=*/false, a, /*neg_b=*/false, bfrag[nt],
                        /*c_mod=*/(short)0, acc[mt][nt],
                        /*reuse_a=*/false, /*reuse_b=*/false);
                }
            }
        }
        __syncthreads();   // next-chunk writes done; this-chunk reads done
    }

    // ---- epilogue: C/D layout -> row = base + (lane>>4)*8 + r, col = lane&15
#pragma unroll
    for (int nt = 0; nt < 2; ++nt) {
        int col = col0 + wave_n * 32 + nt * 16 + lm;
        float bv = bias[col];
#pragma unroll
        for (int mt = 0; mt < 4; ++mt) {
            int rbase = row0 + wave_m * 64 + mt * 16 + lk * 8;
#pragma unroll
            for (int r = 0; r < 8; ++r) {
                out[(size_t)(rbase + r) * OUT_D + col] = acc[mt][nt][r] + bv;
            }
        }
    }
}

// ---------------------------------------------------------------------------
extern "C" void kernel_launch(void* const* d_in, const int* in_sizes, int n_in,
                              void* d_out, int out_size, void* d_ws, size_t ws_size,
                              hipStream_t stream) {
    const float* x    = (const float*)d_in[0];   // [4, 2048, 512] f32
    const float* fc   = (const float*)d_in[1];   // [2, 512, 512, 8] f32
    const float* bias = (const float*)d_in[2];   // [1, 512] f32
    float* out        = (float*)d_out;           // [8192, 512] f32
    _Float16* Wt      = (_Float16*)d_ws;         // 8 MB f16 transposed weights

    fkan_prep_weights<<<(OUT_D * IN_D) / 256, 256, 0, stream>>>(fc, Wt);

    dim3 grid(OUT_D / 128, N_ROWS / 128);        // 4 x 64 workgroups
    fkan_wmma<<<grid, 256, 0, stream>>>(x, Wt, bias, out);
}
